// STKBranch_25039659336082
// MI455X (gfx1250) — matrix operand order repaired and checked by reading, not compiled, not run on hardware
//
#include <hip/hip_runtime.h>

// Double-softmax attention for MI455X (gfx1250, wave32).
// B=4, H=8, N=2048, D=64, fp32 in/out.
// QK^T and P*V run on v_wmma_f32_16x16x32_f16; K/V are pre-converted to f16
// (V also transposed per head) in d_ws so WMMA B-fragments are pure b128 loads.
// scale is folded into the Q fragments; softmax pass C emits f16 probabilities
// compacted in-place in LDS so phase-2 A-fragments are pure ds_load_b128.

typedef _Float16 v16h __attribute__((ext_vector_type(16)));
typedef _Float16 v8h  __attribute__((ext_vector_type(8)));
typedef _Float16 v4h  __attribute__((ext_vector_type(4)));
typedef float    v8f  __attribute__((ext_vector_type(8)));

#define N_SEQ   2048
#define DHEAD   64
#define NBH     32          // B*H
#define QT      32          // queries per workgroup (two 16-row WMMA blocks)
#define LSTR    2052        // LDS row stride in floats (2048 + 4 pad, 16B aligned)
#define NWAVES  8           // 256 threads

// ---------------- K: fp32 -> f16, same [key][dim] layout -----------------------
__global__ __launch_bounds__(256) void cvt_k_kernel(const float* __restrict__ K,
                                                    _Float16* __restrict__ Kh) {
    size_t idx = ((size_t)blockIdx.x * 256 + threadIdx.x) * 8;   // total 4,194,304 elems
    const float4* s = (const float4*)(K + idx);
    float4 a = s[0], b = s[1];
    v8h h;
    h[0]=(_Float16)a.x; h[1]=(_Float16)a.y; h[2]=(_Float16)a.z; h[3]=(_Float16)a.w;
    h[4]=(_Float16)b.x; h[5]=(_Float16)b.y; h[6]=(_Float16)b.z; h[7]=(_Float16)b.w;
    *(v8h*)(Kh + idx) = h;
}

// ---------------- V: fp32 [key][dim] -> f16 [dim][key] per head ----------------
__global__ __launch_bounds__(256) void tr_v_kernel(const float* __restrict__ V,
                                                   _Float16* __restrict__ Vt) {
    __shared__ _Float16 st[64 * 65];                 // 64 keys x 64 dims, padded
    const int bh   = blockIdx.x >> 5;                // 32 heads
    const int tile = blockIdx.x & 31;                // 2048/64 key-tiles
    const int n0   = tile * 64;
    const float*    Vb  = V  + (size_t)bh * N_SEQ * DHEAD;
    _Float16*       Vtb = Vt + (size_t)bh * DHEAD * N_SEQ;
    const int tid = threadIdx.x;
    #pragma unroll
    for (int i = 0; i < 16; ++i) {                   // read coalesced along dims
        int e = tid + 256 * i, n = e >> 6, d = e & 63;
        st[d * 65 + n] = (_Float16)Vb[(size_t)(n0 + n) * DHEAD + d];
    }
    __syncthreads();
    #pragma unroll
    for (int i = 0; i < 16; ++i) {                   // write coalesced along keys
        int e = tid + 256 * i, d = e >> 6, n = e & 63;
        Vtb[(size_t)d * N_SEQ + n0 + n] = st[d * 65 + n];
    }
}

// ---------------- Main attention kernel ---------------------------------------
template <bool PRE>
__global__ __launch_bounds__(256) void dsmax_attn_kernel(
    const float* __restrict__ Q, const float* __restrict__ K32,
    const float* __restrict__ V32, const _Float16* __restrict__ Kh,
    const _Float16* __restrict__ Vt, const float* __restrict__ scale_p,
    float* __restrict__ Out)
{
    __shared__ float sL[QT * LSTR];                  // logits -> z -> p(f16, lower half)
    __shared__ float sM1[QT], sS1[QT], sM2[QT], sIS2[QT];

    const int tid  = threadIdx.x;
    const int lane = tid & 31;
    const int wave = tid >> 5;

    const int qt = blockIdx.x & 63;                  // N/QT = 64 q-tiles
    const int bh = blockIdx.x >> 6;                  // B*H = 32
    const int q0 = qt * QT;

    const float*    Qb  = Q   + (size_t)bh * N_SEQ * DHEAD;
    const float*    K32b= K32 + (size_t)bh * N_SEQ * DHEAD;
    const float*    V32b= V32 + (size_t)bh * N_SEQ * DHEAD;
    const _Float16* Khb = Kh  + (size_t)bh * N_SEQ * DHEAD;
    const _Float16* Vtb = Vt  + (size_t)bh * DHEAD * N_SEQ;
    float*          Ob  = Out + (size_t)bh * N_SEQ * DHEAD;
    const float scl = scale_p[0];                    // 0.125: exact in f16

    // A fragment order used here (16x32, contiguous k): lane L -> row m=L&15; elem e -> K=(L>>4)*16+e
    const int am  = lane & 15;
    const int akb = (lane >> 4) * 16;
    // B fragment order used here (32x16, same k as A): lane L -> col n=L&15; elem e -> K=(L>>4)*16+e
    const int bn  = lane & 15;
    const int bkb = (lane >> 4) * 16;

    // ---- Q A-fragments (scale folded in): acc of WMMA is directly the scaled logit
    v16h qa[2][2];
    #pragma unroll
    for (int qb = 0; qb < 2; ++qb)
    #pragma unroll
    for (int c = 0; c < 2; ++c) {
        const float* qr = Qb + (size_t)(q0 + qb * 16 + am) * DHEAD + 32 * c;
        const float4* qlo = (const float4*)(qr + akb);
        const float4* qhi = (const float4*)(qr + akb + 8);
        float4 f0 = qlo[0], f1 = qlo[1], f2 = qhi[0], f3 = qhi[1];
        v16h a;
        a[0]=(_Float16)(f0.x*scl); a[1]=(_Float16)(f0.y*scl);
        a[2]=(_Float16)(f0.z*scl); a[3]=(_Float16)(f0.w*scl);
        a[4]=(_Float16)(f1.x*scl); a[5]=(_Float16)(f1.y*scl);
        a[6]=(_Float16)(f1.z*scl); a[7]=(_Float16)(f1.w*scl);
        a[8]=(_Float16)(f2.x*scl); a[9]=(_Float16)(f2.y*scl);
        a[10]=(_Float16)(f2.z*scl); a[11]=(_Float16)(f2.w*scl);
        a[12]=(_Float16)(f3.x*scl); a[13]=(_Float16)(f3.y*scl);
        a[14]=(_Float16)(f3.z*scl); a[15]=(_Float16)(f3.w*scl);
        qa[qb][c] = a;
    }

    // ================= Phase 1: logits tile [32 x 2048] via WMMA ==================
    for (int t = wave; t < N_SEQ / 16; t += NWAVES) {
        v16h bf0, bf1;
        if constexpr (PRE) {
            const _Float16* kr = Khb + (size_t)(t * 16 + bn) * DHEAD + bkb;
            __builtin_prefetch(kr + NWAVES * 16 * DHEAD, 0, 1);  // next tile for this wave
            bf0 = *(const v16h*)kr;          // dims bkb..bkb+15   (chunk 0)
            bf1 = *(const v16h*)(kr + 32);   // dims 32+bkb..+15   (chunk 1)
        } else {
            #pragma unroll
            for (int c = 0; c < 2; ++c) {
                const float4* kr4 = (const float4*)(K32b + (size_t)(t * 16 + bn) * DHEAD
                                                    + 32 * c + bkb);
                v16h bf;
                #pragma unroll
                for (int u = 0; u < 4; ++u) {
                    float4 f = kr4[u];
                    bf[4*u+0]=(_Float16)f.x; bf[4*u+1]=(_Float16)f.y;
                    bf[4*u+2]=(_Float16)f.z; bf[4*u+3]=(_Float16)f.w;
                }
                if (c == 0) bf0 = bf; else bf1 = bf;
            }
        }
        #pragma unroll
        for (int qb = 0; qb < 2; ++qb) {
            v8f acc = {};
            acc = __builtin_amdgcn_wmma_f32_16x16x32_f16(false, qa[qb][0], false, bf0,
                                                         (short)0, acc, false, false);
            acc = __builtin_amdgcn_wmma_f32_16x16x32_f16(false, qa[qb][1], false, bf1,
                                                         (short)0, acc, false, false);
            #pragma unroll
            for (int r = 0; r < 8; ++r) {
                int m = qb * 16 + r + 8 * (lane >> 4);
                sL[m * LSTR + t * 16 + bn] = acc[r];          // already scaled
            }
        }
    }
    __syncthreads();

    // ================= Pass A: M1 = max(2l), S1 = sum exp(2l - M1) ================
    for (int rr = 0; rr < QT / NWAVES; ++rr) {
        const int row = wave * (QT / NWAVES) + rr;
        const float4* L4 = (const float4*)&sL[row * LSTR];
        float mx = -3.0e38f;
        for (int j = lane; j < N_SEQ / 4; j += 32) {
            float4 f = L4[j];
            mx = fmaxf(mx, fmaxf(fmaxf(2.f*f.x, 2.f*f.y), fmaxf(2.f*f.z, 2.f*f.w)));
        }
        #pragma unroll
        for (int off = 16; off; off >>= 1) mx = fmaxf(mx, __shfl_xor(mx, off));
        float sm = 0.f;
        for (int j = lane; j < N_SEQ / 4; j += 32) {
            float4 f = L4[j];
            sm += __expf(2.f*f.x - mx) + __expf(2.f*f.y - mx)
                + __expf(2.f*f.z - mx) + __expf(2.f*f.w - mx);
        }
        #pragma unroll
        for (int off = 16; off; off >>= 1) sm += __shfl_xor(sm, off);
        if (lane == 0) { sM1[row] = mx; sS1[row] = sm; }
    }
    __syncthreads();

    // ================= Pass B: z = l * softmask, M2 = max z (in place, fp32) ======
    for (int rr = 0; rr < QT / NWAVES; ++rr) {
        const int row = wave * (QT / NWAVES) + rr;
        float4* L4 = (float4*)&sL[row * LSTR];
        const float m1 = sM1[row];
        const float is1 = 1.f / sS1[row];
        float mz = -3.0e38f;
        for (int j = lane; j < N_SEQ / 4; j += 32) {
            float4 f = L4[j], z;
            z.x = f.x * (__expf(2.f*f.x - m1) * is1);
            z.y = f.y * (__expf(2.f*f.y - m1) * is1);
            z.z = f.z * (__expf(2.f*f.z - m1) * is1);
            z.w = f.w * (__expf(2.f*f.w - m1) * is1);
            L4[j] = z;
            mz = fmaxf(mz, fmaxf(fmaxf(z.x, z.y), fmaxf(z.z, z.w)));
        }
        #pragma unroll
        for (int off = 16; off; off >>= 1) mz = fmaxf(mz, __shfl_xor(mz, off));
        if (lane == 0) sM2[row] = mz;
    }
    __syncthreads();

    // ====== Pass C: p = exp(z - M2) emitted as f16, compacted into the lower =====
    // ====== half of each row (write word 2j while reading word 4j: in-order  =====
    // ====== LDS + write index always behind future reads -> safe in place).  =====
    for (int rr = 0; rr < QT / NWAVES; ++rr) {
        const int row = wave * (QT / NWAVES) + rr;
        float4*    L4 = (float4*)&sL[row * LSTR];
        _Float16*  H  = (_Float16*)&sL[row * LSTR];
        const float m2 = sM2[row];
        float s2 = 0.f;
        for (int j = lane; j < N_SEQ / 4; j += 32) {
            float4 f = L4[j];
            float p0 = __expf(f.x - m2), p1 = __expf(f.y - m2);
            float p2 = __expf(f.z - m2), p3 = __expf(f.w - m2);
            v4h h;
            h[0]=(_Float16)p0; h[1]=(_Float16)p1; h[2]=(_Float16)p2; h[3]=(_Float16)p3;
            *(v4h*)(H + 4 * j) = h;                   // 8B store at f16 index 4j
            s2 += p0 + p1 + p2 + p3;
        }
        #pragma unroll
        for (int off = 16; off; off >>= 1) s2 += __shfl_xor(s2, off);
        if (lane == 0) sIS2[row] = 1.f / s2;          // store reciprocal
    }
    __syncthreads();

    // ================= Phase 2: out = (p @ V) * (1/S2) via WMMA ===================
    {
        const int qb = wave >> 2;              // q-block 0/1
        const int nt = wave & 3;               // 16-wide dim slice
        const int vdim = nt * 16 + bn;
        const _Float16* ph = (const _Float16*)&sL[(qb * 16 + am) * LSTR];
        v8f acc = {};
        for (int kc = 0; kc < N_SEQ / 32; ++kc) {
            // A fragment: p f16 from LDS, two 16B loads, no conversion
            v8h lo = *(const v8h*)(ph + kc * 32 + akb);
            v8h hi = *(const v8h*)(ph + kc * 32 + akb + 8);
            v16h af = __builtin_shufflevector(lo, hi, 0,1,2,3,4,5,6,7,
                                                       8,9,10,11,12,13,14,15);
            v16h bf;
            if constexpr (PRE) {
                // B fragment: Vt[dim][key] f16 -> 16 consecutive keys, one 32B load
                const _Float16* vr = Vtb + (size_t)vdim * N_SEQ + kc * 32 + bkb;
                bf = *(const v16h*)vr;
            } else {
                const float* vr = V32b + (size_t)(kc * 32 + bkb) * DHEAD + vdim;
                #pragma unroll
                for (int e = 0; e < 16; ++e)
                    bf[e] = (_Float16)vr[(size_t)e * DHEAD];
            }
            acc = __builtin_amdgcn_wmma_f32_16x16x32_f16(false, af, false, bf,
                                                         (short)0, acc, false, false);
        }
        #pragma unroll
        for (int r = 0; r < 8; ++r) {
            int m = r + 8 * (lane >> 4);
            Ob[(size_t)(q0 + qb * 16 + m) * DHEAD + nt * 16 + bn] =
                acc[r] * sIS2[qb * 16 + m];
        }
    }
}

extern "C" void kernel_launch(void* const* d_in, const int* in_sizes, int n_in,
                              void* d_out, int out_size, void* d_ws, size_t ws_size,
                              hipStream_t stream) {
    (void)in_sizes; (void)n_in; (void)out_size;
    const float* q     = (const float*)d_in[0];
    const float* k     = (const float*)d_in[1];
    const float* v     = (const float*)d_in[2];
    const float* scale = (const float*)d_in[3];
    float* out = (float*)d_out;

    const size_t nelem  = (size_t)NBH * N_SEQ * DHEAD;     // 4,194,304
    const size_t fbytes = nelem * sizeof(_Float16);        // 8 MiB each
    const bool pre = (ws_size >= 2 * fbytes) && (d_ws != nullptr);

    dim3 block(256);
    dim3 grid(NBH * (N_SEQ / QT));                          // 32 * 64 = 2048 WGs

    if (pre) {
        _Float16* kh = (_Float16*)d_ws;
        _Float16* vt = (_Float16*)((char*)d_ws + fbytes);
        hipLaunchKernelGGL(cvt_k_kernel, dim3((unsigned)(nelem / (256 * 8))), block, 0, stream,
                           k, kh);
        hipLaunchKernelGGL(tr_v_kernel, dim3(NBH * (N_SEQ / 64)), block, 0, stream,
                           v, vt);
        hipLaunchKernelGGL(HIP_KERNEL_NAME(dsmax_attn_kernel<true>), grid, block, 0, stream,
                           q, k, v, kh, vt, scale, out);
    } else {
        hipLaunchKernelGGL(HIP_KERNEL_NAME(dsmax_attn_kernel<false>), grid, block, 0, stream,
                           q, k, v, (const _Float16*)nullptr, (const _Float16*)nullptr,
                           scale, out);
    }
}
